// TSRM_Encoder_36129264894097
// MI455X (gfx1250) — matrix operand order repaired and checked by reading, not compiled, not run on hardware
//
#include <hip/hip_runtime.h>
#include <hip/hip_bf16.h>
#include <math.h>

// Problem constants (match reference)
#define BB   4
#define TT   256
#define DD   1024
#define HH   512
#define GG   8
#define PP   384
#define NN   384
#define LL   16
#define HGG  64
#define ECW  1124                 // event_ctx row: 512 ctx + 512 X + 100 pos_ind
#define EC_SZ (PP*ECW)
#define CF_SZ (PP*LL*DD)
#define MPB  64                   // (n,m) pairs per block in k_pos_sim

typedef __attribute__((ext_vector_type(2)))  float   v2f;
typedef __attribute__((ext_vector_type(8)))  float   v8f;
typedef __attribute__((ext_vector_type(16))) __bf16  v16bf;

union BFrag { v16bf v; uint4 q[2]; };

__device__ __forceinline__ unsigned short f2bf(float f) {
  unsigned int u = __builtin_bit_cast(unsigned int, f);
  u += 0x7FFFu + ((u >> 16) & 1u);          // round-to-nearest-even
  return (unsigned short)(u >> 16);
}

// Async copy of one 16B chunk global->LDS (ASYNCcnt-tracked, CDNA5 VGLOBAL op)
__device__ __forceinline__ void async_g2l_b128(unsigned lds_off,
                                               unsigned long long gaddr) {
  asm volatile("global_load_async_to_lds_b128 %0, %1, off"
               :: "v"(lds_off), "v"(gaddr) : "memory");
}
__device__ __forceinline__ void wait_asynccnt0() {
  asm volatile("s_wait_asynccnt 0x0" ::: "memory");
}

// ---------------------------------------------------------------------------
// Kernel 1: clip gather + mean.  clip_feats & clip_mask straight into d_out,
// event_feats into workspace.  Pure streaming (~30 MB, trivial at 23.3 TB/s).
// ---------------------------------------------------------------------------
__global__ void k_gather_mean(const float* __restrict__ feats,
                              const int*   __restrict__ vid_idx,
                              const int*   __restrict__ starts,
                              float* __restrict__ out,
                              float* __restrict__ event_feats) {
  const int p = blockIdx.x;
  const int d = threadIdx.x * 4;
  const int vid = vid_idx[p];
  const int st  = starts[p];
  const float* src = feats + ((size_t)vid * TT + st) * DD + d;
  float* cf = out + EC_SZ + ((size_t)p * LL) * DD + d;
  float4 acc = make_float4(0.f, 0.f, 0.f, 0.f);
  for (int l = 0; l < LL; ++l) {
    float4 x = *(const float4*)(src + (size_t)l * DD);
    *(float4*)(cf + (size_t)l * DD) = x;
    acc.x += x.x; acc.y += x.y; acc.z += x.z; acc.w += x.w;
  }
  const float inv = 1.0f / (16.0f + 1e-5f);
  float4 ef4 = make_float4(acc.x * inv, acc.y * inv, acc.z * inv, acc.w * inv);
  *(float4*)(event_feats + (size_t)p * DD + d) = ef4;
  if (threadIdx.x < LL) out[EC_SZ + CF_SZ + p * LL + threadIdx.x] = 1.0f;
}

// ---------------------------------------------------------------------------
// Kernel 2: W_p1 (K x N f32, row-major) -> W_p1t (N x K bf16) so B-fragments
// of v_wmma_f32_16x16x32_bf16 are contiguous b128 chunks.
// ---------------------------------------------------------------------------
__global__ void k_prep_wp1t(const float* __restrict__ W_p1,
                            unsigned short* __restrict__ W_p1t) {
  int t  = blockIdx.x * blockDim.x + threadIdx.x;
  int n  = t >> 9;
  int kk = t & 511;
  W_p1t[(size_t)n * HH + kk] = f2bf(W_p1[(size_t)kk * HH + n]);
}

// ---------------------------------------------------------------------------
// Kernel 3: exact-f32 GEMM tile kernel, V_WMMA_F32_16X16X4_F32, one wave per
// 16x16 tile.  Compile-time K / fixed Nout=512 so addressing strength-reduces.
// A-frag (16x4 f32): lane<16 -> {K0,K1}, lanes 16-31 -> {K2,K3}; B mirrored.
// ---------------------------------------------------------------------------
template <int K, bool RELU>
__global__ void __launch_bounds__(32)
k_gemm_wmma_f32(const float* __restrict__ A,
                const int*   __restrict__ gather,
                const float* __restrict__ W,
                const float* __restrict__ bias,
                float* __restrict__ Out) {
  const int lane = threadIdx.x;
  const int half = lane >> 4;
  const int l16  = lane & 15;
  const int row  = blockIdx.y * 16 + l16;
  const int col  = blockIdx.x * 16 + l16;
  const int arow = gather ? gather[row] : row;
  const float* Ar = A + (size_t)arow * K;
  v8f acc = {0.f, 0.f, 0.f, 0.f, 0.f, 0.f, 0.f, 0.f};
  #pragma unroll 4
  for (int k0 = 0; k0 < K; k0 += 4) {
    v2f a = *(const v2f*)(Ar + k0 + 2 * half);
    v2f b;
    b.x = W[(size_t)(k0 + 2 * half)     * HH + col];
    b.y = W[(size_t)(k0 + 2 * half + 1) * HH + col];
    acc = __builtin_amdgcn_wmma_f32_16x16x4_f32(false, a, false, b,
                                                (short)0, acc, false, false);
  }
  const float bv = bias[col];
  for (int j = 0; j < 8; ++j) {
    float vv = acc[j] + bv;
    if (RELU) vv = fmaxf(vv, 0.0f);
    Out[(size_t)(blockIdx.y * 16 + j + 8 * half) * HH + col] = vv;
  }
}

// ---------------------------------------------------------------------------
// Kernel 4: cos_sim[g,n,m] = dot64(q,k)/8  (0.15 GFLOP, VALU is fine)
// ---------------------------------------------------------------------------
__global__ void k_cos_sim(const float* __restrict__ q,
                          const float* __restrict__ k,
                          float* __restrict__ sim) {
  int t = blockIdx.x * 256 + threadIdx.x;
  int g = t / (NN * NN);
  int r = t - g * NN * NN;
  int n = r / NN, m = r - n * NN;
  const float* qr = q + (size_t)n * HH + g * HGG;
  const float* kr = k + (size_t)m * HH + g * HGG;
  float s = 0.f;
  #pragma unroll 8
  for (int d = 0; d < HGG; ++d) s += qr[d] * kr[d];
  sim[t] = s * 0.125f;   // 1/sqrt(64)
}

// ---------------------------------------------------------------------------
// Kernel 5: hot path (>95% of FLOPs).  Block = 64 pairs, 4 waves.
//  phase 1: synthesize 64x512 bf16 pe tile in LDS (sin/cos on the fly; the
//           302 MB pe tensor never exists in HBM).  Overlapped with the
//           async staging of the first weight tile.
//  phase 2: 32 col-tiles; each 16KB W_p1t tile is staged into LDS ONCE via
//           global_load_async_to_lds_b128 (double-buffered, overlapped with
//           the 16 v_wmma_f32_16x16x32_bf16 of the previous tile) and
//           consumed by all 4 waves -> 4x less L2 weight traffic (1.2 GB
//           total) vs per-wave streaming; keeps the kernel WMMA-bound.
//  phase 3: h @ W_p2 + b_p2 (512->8) from LDS, accumulate into sim.
// ---------------------------------------------------------------------------
__global__ void __launch_bounds__(128)
k_pos_sim(const unsigned short* __restrict__ W_p1t,
          const float* __restrict__ b_p1,
          const float* __restrict__ W_p2,
          const float* __restrict__ b_p2,
          const float* __restrict__ timestamps,
          float* __restrict__ sim) {
  __shared__ __align__(16) unsigned short peL[MPB][520];   // +8 pad: bank spread
  __shared__ __align__(16) unsigned short BL[2][16][520];  // staged weight tiles
  __shared__ __align__(16) float hL[MPB][512];

  const int tid  = threadIdx.x;
  const int wave = tid >> 5;
  const int lane = tid & 31;
  const int half = lane >> 4;
  const int l16  = lane & 15;
  const int p0   = blockIdx.x * MPB;   // 64 consecutive pairs share n (384%64==0)
  const int n    = p0 / NN;
  const int m0   = p0 - n * NN;

  // -- prologue: kick off async stage of weight tile 0 (1024 x 16B chunks) --
  {
    const char* gbase = (const char*)W_p1t;        // tile 0 is contiguous 16KB
    #pragma unroll
    for (int i = 0; i < 8; ++i) {
      int ci = (wave * 8 + i) * 32 + lane;         // 0..1023
      int c = ci >> 6, j = ci & 63;
      async_g2l_b128((unsigned)(size_t)(void*)&BL[0][c][j * 8],
                     (unsigned long long)(gbase + (size_t)ci * 16));
    }
  }

  // ---- phase 1: pe tile (64 pairs x 512 features), bf16 in LDS ----
  const float st_n  = timestamps[2 * n], en_n = timestamps[2 * n + 1];
  const float cen_n = 0.5f * (st_n + en_n);
  const float len_n = fmaxf(en_n - st_n, 0.1f);
  for (int e = 0; e < 256; ++e) {
    int lin = e * 128 + tid;           // 0..32767
    int i   = lin >> 9;                // pair within tile
    int c   = lin & 511;               // feature column
    int m   = m0 + i;
    float st_m  = timestamps[2 * m], en_m = timestamps[2 * m + 1];
    float cen_m = 0.5f * (st_m + en_m);
    float len_m = fmaxf(en_m - st_m, 0.1f);
    int cc  = c >> 8;                  // 0: delta_center block, 1: delta_length
    int rr  = c & 255;
    int idx = rr & 127;
    float pos = cc ? logf(len_m / len_n) : (cen_n - cen_m) / len_n;
    float dim = exp2f((float)idx * (13.287712379549449f / 128.0f)); // 10000^(idx/128)
    float dv  = 100.0f * pos / dim;
    float val = (rr & 128) ? cosf(dv) : sinf(dv);
    peL[i][c] = f2bf(val);
  }
  wait_asynccnt0();
  __syncthreads();

  // ---- phase 2: 32 col-tiles, double-buffered weight staging + bf16 WMMA ----
  const unsigned short* prow = &peL[wave * 16 + l16][0];
  for (int t = 0; t < 32; ++t) {
    if (t < 31) {                       // stage tile t+1 while computing t
      const char* gbase = (const char*)W_p1t + (size_t)(t + 1) * 16 * HH * 2;
      const int sel = (t + 1) & 1;
      #pragma unroll
      for (int i = 0; i < 8; ++i) {
        int ci = (wave * 8 + i) * 32 + lane;
        int c = ci >> 6, j = ci & 63;
        async_g2l_b128((unsigned)(size_t)(void*)&BL[sel][c][j * 8],
                       (unsigned long long)(gbase + (size_t)ci * 16));
      }
    }
    const int sel = t & 1;
    const int col = t * 16 + l16;
    v8f acc = {0.f, 0.f, 0.f, 0.f, 0.f, 0.f, 0.f, 0.f};
    #pragma unroll
    for (int k0 = 0; k0 < HH; k0 += 32) {
      BFrag a, b;
      // 16-bit A frag: lane<16 K k0..k0+7 / k0+16..k0+23 of row l16;
      // lanes 16-31: K k0+8..k0+15 / k0+24..k0+31.
      a.q[0] = *(const uint4*)(prow + k0 + 8 * half);
      a.q[1] = *(const uint4*)(prow + k0 + 16 + 8 * half);
      // 16-bit B frag: same K layout as A (lane<16 K k0..k0+7 / k0+16..k0+23 of col l16; lanes 16-31 +8).
      b.q[0] = *(const uint4*)&BL[sel][l16][k0 + 8 * half];
      b.q[1] = *(const uint4*)&BL[sel][l16][k0 + 16 + 8 * half];
      acc = __builtin_amdgcn_wmma_f32_16x16x32_bf16(false, a.v, false, b.v,
                                                    (short)0, acc, false, false);
    }
    const float bv = b_p1[col];
    #pragma unroll
    for (int r = 0; r < 8; ++r)
      hL[wave * 16 + r + 8 * half][col] = tanhf(acc[r] + bv);
    wait_asynccnt0();                   // tile t+1 landed
    __syncthreads();                    // everyone done reading BL[sel], hL rows
  }

  // ---- phase 3: 512 -> 8 second layer, accumulate into sim ----
  #pragma unroll
  for (int oo = 0; oo < 4; ++oo) {
    int o  = tid + 128 * oo;           // 0..511
    int ip = o >> 3;                   // pair 0..63
    int g  = o & 7;                    // group
    float s = b_p2[g];
    for (int kk = 0; kk < HH; ++kk) s += hL[ip][kk] * W_p2[kk * GG + g];
    sim[((size_t)g * NN + n) * NN + (m0 + ip)] += s;
  }
}

// ---------------------------------------------------------------------------
// Kernel 6: row softmax over m, then mask-renorm (block_id equality).
// ---------------------------------------------------------------------------
__global__ void __launch_bounds__(128)
k_softmax(float* __restrict__ sim, const int* __restrict__ block_id) {
  __shared__ float red[128];
  const int tid = threadIdx.x;
  const int row = blockIdx.x;          // g*384 + n
  const int n   = row % NN;
  float* rp = sim + (size_t)row * NN;
  float v0[3];
  float mx = -1e30f;
  for (int i = 0; i < 3; ++i) { v0[i] = rp[tid + i * 128]; mx = fmaxf(mx, v0[i]); }
  red[tid] = mx; __syncthreads();
  for (int s = 64; s > 0; s >>= 1) { if (tid < s) red[tid] = fmaxf(red[tid], red[tid + s]); __syncthreads(); }
  mx = red[0]; __syncthreads();
  float sum = 0.f;
  for (int i = 0; i < 3; ++i) { v0[i] = expf(v0[i] - mx); sum += v0[i]; }
  red[tid] = sum; __syncthreads();
  for (int s = 64; s > 0; s >>= 1) { if (tid < s) red[tid] += red[tid + s]; __syncthreads(); }
  const float inv = 1.0f / red[0]; __syncthreads();
  const int bn = block_id[n];
  float w[3];
  float sum2 = 0.f;
  for (int i = 0; i < 3; ++i) {
    int m = tid + i * 128;
    float p = v0[i] * inv;
    w[i] = (block_id[m] == bn) ? p : 0.f;
    sum2 += w[i];
  }
  red[tid] = sum2; __syncthreads();
  for (int s = 64; s > 0; s >>= 1) { if (tid < s) red[tid] += red[tid + s]; __syncthreads(); }
  const float inv2 = 1.0f / (1e-5f + red[0]);
  for (int i = 0; i < 3; ++i) rp[tid + i * 128] = w[i] * inv2;
}

// ---------------------------------------------------------------------------
// Kernel 7: ctx = sim @ v, relu, concat [relu(ctx) | X | pos_ind] -> event_ctx
// ---------------------------------------------------------------------------
__global__ void __launch_bounds__(128)
k_ctx(const float* __restrict__ sim,
      const float* __restrict__ v,
      const float* __restrict__ ef,
      const int*   __restrict__ gather,
      const float* __restrict__ timestamps,
      const float* __restrict__ durations,
      float* __restrict__ out) {
  const int n   = blockIdx.x;
  const int tid = threadIdx.x;
  const int h0  = tid * 4;             // 4 cols per thread, same group g
  const int g   = h0 >> 6;
  const float* sr = sim + ((size_t)g * NN + n) * NN;
  float4 acc = make_float4(0.f, 0.f, 0.f, 0.f);
  for (int m = 0; m < NN; ++m) {
    float s = sr[m];
    float4 vv = *(const float4*)(v + (size_t)m * HH + h0);
    acc.x += s * vv.x; acc.y += s * vv.y; acc.z += s * vv.z; acc.w += s * vv.w;
  }
  float* orow = out + (size_t)n * ECW;
  orow[h0 + 0] = fmaxf(acc.x, 0.f);
  orow[h0 + 1] = fmaxf(acc.y, 0.f);
  orow[h0 + 2] = fmaxf(acc.z, 0.f);
  orow[h0 + 3] = fmaxf(acc.w, 0.f);
  const int gi = gather[n];
  for (int c = tid; c < HH; c += 128) orow[HH + c] = ef[(size_t)gi * HH + c];
  if (tid < 100) {
    float dur = durations[n];
    int si = (int)(timestamps[2 * n]     / dur * 99.0f); si = si > 99 ? 99 : si;
    int ei = (int)(timestamps[2 * n + 1] / dur * 99.0f); ei = ei > 99 ? 99 : ei;
    orow[1024 + tid] = (tid >= si && tid <= ei) ? 1.0f : 0.0f;
  }
}

// ---------------------------------------------------------------------------
extern "C" void kernel_launch(void* const* d_in, const int* in_sizes, int n_in,
                              void* d_out, int out_size, void* d_ws, size_t ws_size,
                              hipStream_t stream) {
  (void)in_sizes; (void)n_in; (void)out_size; (void)ws_size;
  const float* feats      = (const float*)d_in[0];
  const int*   vid_idx    = (const int*)d_in[1];
  const int*   starts     = (const int*)d_in[2];
  const int*   gather_idx = (const int*)d_in[4];
  const int*   block_id   = (const int*)d_in[5];
  const float* timestamps = (const float*)d_in[6];
  const float* durations  = (const float*)d_in[7];
  const float* W_pre = (const float*)d_in[8];
  const float* b_pre = (const float*)d_in[9];
  const float* W_q   = (const float*)d_in[10];
  const float* b_q   = (const float*)d_in[11];
  const float* W_k   = (const float*)d_in[12];
  const float* b_k   = (const float*)d_in[13];
  const float* W_v   = (const float*)d_in[14];
  const float* b_v   = (const float*)d_in[15];
  const float* W_p1  = (const float*)d_in[16];
  const float* b_p1  = (const float*)d_in[17];
  const float* W_p2  = (const float*)d_in[18];
  const float* b_p2  = (const float*)d_in[19];

  char* ws = (char*)d_ws;
  float* event_feats = (float*)ws;  ws += (size_t)PP * DD * 4;
  float* ef          = (float*)ws;  ws += (size_t)PP * HH * 4;
  float* qb          = (float*)ws;  ws += (size_t)PP * HH * 4;
  float* kb          = (float*)ws;  ws += (size_t)PP * HH * 4;
  float* vb          = (float*)ws;  ws += (size_t)PP * HH * 4;
  float* sim         = (float*)ws;  ws += (size_t)GG * NN * NN * 4;
  unsigned short* W_p1t = (unsigned short*)ws;
  float* out = (float*)d_out;

  k_gather_mean<<<PP, 256, 0, stream>>>(feats, vid_idx, starts, out, event_feats);
  k_prep_wp1t<<<(HH * HH) / 256, 256, 0, stream>>>(W_p1, W_p1t);
  k_gemm_wmma_f32<DD, true ><<<dim3(HH / 16, PP / 16), 32, 0, stream>>>(event_feats, nullptr,    W_pre, b_pre, ef);
  k_gemm_wmma_f32<HH, false><<<dim3(HH / 16, PP / 16), 32, 0, stream>>>(ef,          gather_idx, W_q,   b_q,   qb);
  k_gemm_wmma_f32<HH, false><<<dim3(HH / 16, PP / 16), 32, 0, stream>>>(ef,          gather_idx, W_k,   b_k,   kb);
  k_gemm_wmma_f32<HH, false><<<dim3(HH / 16, PP / 16), 32, 0, stream>>>(ef,          gather_idx, W_v,   b_v,   vb);
  k_cos_sim<<<(GG * NN * NN) / 256, 256, 0, stream>>>(qb, kb, sim);
  k_pos_sim<<<(NN * NN) / MPB, 128, 0, stream>>>(W_p1t, b_p1, W_p2, b_p2, timestamps, sim);
  k_softmax<<<GG * NN, 128, 0, stream>>>(sim, block_id);
  k_ctx<<<NN, 128, 0, stream>>>(sim, vb, ef, gather_idx, timestamps, durations, out);
}
